// ADCtoRAMapFull_79465484910746
// MI455X (gfx1250) — matrix operand order repaired and checked by reading, not compiled, not run on hardware
//
#include <hip/hip_runtime.h>
#include <math.h>

// ---------------- problem constants ----------------
#define R_SAMP 512
#define D_CHRP 256
#define NRX    16
#define A_BINS 751
#define NTX    12
#define VDIM   192          // NTX*NRX
#define NPIX   (R_SAMP*D_CHRP)   // 131072
#define A_PAD  752          // 47*16

typedef __attribute__((ext_vector_type(16))) _Float16 v16h;
typedef __attribute__((ext_vector_type(8)))  float    v8f;

// ---------------- stage 0: per-channel mean ----------------
__global__ __launch_bounds__(256) void mean_kernel(const float* __restrict__ x,
                                                   float* __restrict__ mean) {
    __shared__ float s[256];
    int c = blockIdx.x;           // 0..31 (re channels 0..15, im 16..31)
    int t = threadIdx.x;
    float acc = 0.f;
    for (int n = t; n < NPIX; n += 256)
        acc += x[(size_t)n * 32 + c];
    s[t] = acc;
    __syncthreads();
    for (int o = 128; o > 0; o >>= 1) {
        if (t < o) s[t] += s[t + o];
        __syncthreads();
    }
    if (t == 0) mean[c] = s[0] * (1.0f / (float)NPIX);
}

// ---------------- stage 1: mean-sub + Hamming + 512-pt range FFT ----------------
// one workgroup per (doppler d, rx) column; LDS radix-2 DIT
__global__ __launch_bounds__(256) void range_fft_kernel(const float* __restrict__ x,
                                                        const float* __restrict__ mean,
                                                        float2* __restrict__ buf1) {
    __shared__ float2 s[R_SAMP];
    int blk = blockIdx.x;              // 0..4095
    int d   = blk >> 4;
    int rx  = blk & 15;
    int t   = threadIdx.x;             // 0..255
    float mre = mean[rx], mim = mean[16 + rx];
    for (int r = t; r < R_SAMP; r += 256) {
        float win = 0.54f - 0.46f * __cosf(6.28318530718f * (float)r / 511.0f);
        const float* px = x + ((size_t)r * D_CHRP + d) * 32;
        float re = (px[rx]      - mre) * win;
        float im = (px[16 + rx] - mim) * win;
        int pos = (int)(__brev((unsigned)r) >> 23);   // bitrev9
        s[pos] = make_float2(re, im);
    }
    __syncthreads();
    for (int st = 0; st < 9; ++st) {                  // 256 butterflies / stage
        int half = 1 << st;
        int j    = t & (half - 1);
        int g    = t >> st;
        int base = g * (half << 1) + j;
        float ang = -6.28318530718f * (float)j / (float)(half << 1);
        float sn, cs;
        __sincosf(ang, &sn, &cs);
        float2 u = s[base];
        float2 w = s[base + half];
        float vr = w.x * cs - w.y * sn;
        float vi = w.x * sn + w.y * cs;
        s[base]        = make_float2(u.x + vr, u.y + vi);
        s[base + half] = make_float2(u.x - vr, u.y - vi);
        __syncthreads();
    }
    for (int k = t; k < R_SAMP; k += 256)             // layout [range][dop][rx]
        buf1[((size_t)k * D_CHRP + d) * NRX + rx] = s[k];
}

// ---------------- stage 2: Doppler window + 256-pt FFT (in place) ----------------
__global__ __launch_bounds__(128) void dop_fft_kernel(float2* __restrict__ buf1) {
    __shared__ float2 s[D_CHRP];
    int blk = blockIdx.x;              // 0..8191
    int r   = blk >> 4;
    int rx  = blk & 15;
    int t   = threadIdx.x;             // 0..127
    for (int i = t; i < D_CHRP; i += 128) {
        float2 v = buf1[((size_t)r * D_CHRP + i) * NRX + rx];
        float win = 0.54f - 0.46f * __cosf(6.28318530718f * (float)i / 255.0f);
        int pos = (int)(__brev((unsigned)i) >> 24);   // bitrev8
        s[pos] = make_float2(v.x * win, v.y * win);
    }
    __syncthreads();
    for (int st = 0; st < 8; ++st) {                  // 128 butterflies / stage
        int half = 1 << st;
        int j    = t & (half - 1);
        int g    = t >> st;
        int base = g * (half << 1) + j;
        float ang = -6.28318530718f * (float)j / (float)(half << 1);
        float sn, cs;
        __sincosf(ang, &sn, &cs);
        float2 u = s[base];
        float2 w = s[base + half];
        float vr = w.x * cs - w.y * sn;
        float vi = w.x * sn + w.y * cs;
        s[base]        = make_float2(u.x + vr, u.y + vi);
        s[base + half] = make_float2(u.x - vr, u.y - vi);
        __syncthreads();
    }
    for (int k = t; k < D_CHRP; k += 128)
        buf1[((size_t)r * D_CHRP + k) * NRX + rx] = s[k];
}

// ---------------- stage 3: Doppler demux + bf_window, f32 -> f16 B panels ----------------
__global__ __launch_bounds__(VDIM) void pack_kernel(const float2* __restrict__ buf1,
                                                    const float* __restrict__ bfwin,
                                                    _Float16* __restrict__ Br,
                                                    _Float16* __restrict__ Bi) {
    int n  = blockIdx.x;               // 0..131071 : n = r*256 + db
    int v  = threadIdx.x;              // 0..191    : v = t*16 + rx
    int r  = n >> 8, db = n & 255;
    int tt = v >> 4, rx = v & 15;
    int jt = (tt == 0) ? 0 : (tt + 4); // kept TX indices {0,5..15}
    int k  = (db + 16 * jt) & 255;     // DOPPLER_IDX[db][tt]
    float2 val = buf1[((size_t)r * D_CHRP + k) * NRX + rx];
    float  w   = bfwin[v];
    Br[(size_t)n * VDIM + v] = (_Float16)(val.x * w);
    Bi[(size_t)n * VDIM + v] = (_Float16)(val.y * w);
}

// ---------------- stage 3b: calib f32 -> f16 A panels (pad to 752 rows) ----------------
__global__ __launch_bounds__(256) void calib_pack_kernel(const float* __restrict__ cr,
                                                         const float* __restrict__ ci,
                                                         _Float16* __restrict__ Ar,
                                                         _Float16* __restrict__ Ai,
                                                         _Float16* __restrict__ mAi) {
    int idx = blockIdx.x * 256 + threadIdx.x;
    if (idx >= A_PAD * VDIM) return;
    int a = idx / VDIM;
    float re = 0.f, im = 0.f;
    if (a < A_BINS) { re = cr[idx]; im = ci[idx]; }
    Ar[idx]  = (_Float16)re;
    Ai[idx]  = (_Float16)im;
    mAi[idx] = (_Float16)(-im);
}

// ---------------- stage 4: complex WMMA GEMM + |.| + transposed store ----------------
// A fragment (16x32 f16): lane L: M=L&15; VGPR v holds K pair
//   v<4 : K0 = 2v + 8*(L>>4) ; v>=4 : K0 = 16 + 2(v-4) + 8*(L>>4)
union V16U { v16h h; unsigned u[8]; };

__device__ inline v16h loadA(const _Float16* __restrict__ p, int lane) {
    V16U x;
    int M   = lane & 15;
    int hi8 = (lane >> 4) << 3;
    const _Float16* row = p + (size_t)M * VDIM;
#pragma unroll
    for (int v = 0; v < 8; ++v) {
        int K0 = (v < 4) ? (2 * v + hi8) : (16 + 2 * (v - 4) + hi8);
        x.u[v] = *(const unsigned*)(row + K0);
    }
    return x.h;
}

// B fragment (32x16 f16): lane L: N=L&15; VGPR v holds K pair, same K order as A
__device__ inline v16h loadB(const _Float16* __restrict__ p, int lane) {
    V16U x;
    int N    = lane & 15;
    int koff = (lane >> 4) << 3;
    const _Float16* row = p + (size_t)N * VDIM;
#pragma unroll
    for (int v = 0; v < 8; ++v)
        x.u[v] = *(const unsigned*)(row + ((v < 4) ? (2 * v + koff) : (16 + 2 * (v - 4) + koff)));
    return x.h;
}

__global__ __launch_bounds__(256) void bf_gemm_kernel(const _Float16* __restrict__ Ar,
                                                      const _Float16* __restrict__ Ai,
                                                      const _Float16* __restrict__ mAi,
                                                      const _Float16* __restrict__ Br,
                                                      const _Float16* __restrict__ Bi,
                                                      float* __restrict__ out) {
    int wave  = threadIdx.x >> 5;
    int lane  = threadIdx.x & 31;
    int atile = blockIdx.y;                      // 0..46
    int ntile = blockIdx.x * 8 + wave;           // 0..8191
    int a0    = atile * 16;
    size_t n0 = (size_t)ntile * 16;

    v8f accRe = {0.f,0.f,0.f,0.f,0.f,0.f,0.f,0.f};
    v8f accIm = {0.f,0.f,0.f,0.f,0.f,0.f,0.f,0.f};

#pragma unroll
    for (int kk = 0; kk < VDIM; kk += 32) {
        v16h ar  = loadA(Ar  + (size_t)a0 * VDIM + kk, lane);
        v16h ai  = loadA(Ai  + (size_t)a0 * VDIM + kk, lane);
        v16h nai = loadA(mAi + (size_t)a0 * VDIM + kk, lane);
        v16h br  = loadB(Br  + n0 * VDIM + kk, lane);
        v16h bi  = loadB(Bi  + n0 * VDIM + kk, lane);
        // Re += Ar*Br - Ai*Bi ; Im += Ar*Bi + Ai*Br
        accRe = __builtin_amdgcn_wmma_f32_16x16x32_f16(false, ar,  false, br, (short)0, accRe, false, false);
        accRe = __builtin_amdgcn_wmma_f32_16x16x32_f16(false, nai, false, bi, (short)0, accRe, false, false);
        accIm = __builtin_amdgcn_wmma_f32_16x16x32_f16(false, ar,  false, bi, (short)0, accIm, false, false);
        accIm = __builtin_amdgcn_wmma_f32_16x16x32_f16(false, ai,  false, br, (short)0, accIm, false, false);
    }

    // C/D layout: VGPR i -> M = i + 8*(lane>=16), N = lane&15
    int hi   = (lane >> 4) << 3;
    int ncol = (int)(n0 + (lane & 15));
    int r = ncol >> 8, d = ncol & 255;
    size_t obase = ((size_t)d * R_SAMP + r) * A_BINS;   // out[d][r][a]
#pragma unroll
    for (int i = 0; i < 8; ++i) {
        int a = a0 + i + hi;
        if (a < A_BINS) {
            float re = accRe[i], im = accIm[i];
            out[obase + a] = sqrtf(re * re + im * im);
        }
    }
}

// ---------------- host launcher ----------------
extern "C" void kernel_launch(void* const* d_in, const int* in_sizes, int n_in,
                              void* d_out, int out_size, void* d_ws, size_t ws_size,
                              hipStream_t stream) {
    const float* x          = (const float*)d_in[0];
    const float* calib_real = (const float*)d_in[1];
    const float* calib_imag = (const float*)d_in[2];
    const float* bf_window  = (const float*)d_in[3];
    float* out = (float*)d_out;

    char* ws = (char*)d_ws;
    // workspace layout (≈118 MB)
    float*    mean = (float*)ws;                                       // 128 B
    float2*   buf1 = (float2*)(ws + 256);                              // 16 MB
    size_t off = 256 + (size_t)NPIX * NRX * sizeof(float2);            // 16,777,472
    _Float16* Br  = (_Float16*)(ws + off);  off += (size_t)NPIX * VDIM * 2;   // 48 MB
    _Float16* Bi  = (_Float16*)(ws + off);  off += (size_t)NPIX * VDIM * 2;   // 48 MB
    _Float16* Ar  = (_Float16*)(ws + off);  off += (size_t)A_PAD * VDIM * 2;
    _Float16* Ai  = (_Float16*)(ws + off);  off += (size_t)A_PAD * VDIM * 2;
    _Float16* mAi = (_Float16*)(ws + off);  off += (size_t)A_PAD * VDIM * 2;
    (void)ws_size; (void)in_sizes; (void)n_in; (void)out_size;

    mean_kernel<<<32, 256, 0, stream>>>(x, mean);
    range_fft_kernel<<<D_CHRP * NRX, 256, 0, stream>>>(x, mean, buf1);
    dop_fft_kernel<<<R_SAMP * NRX, 128, 0, stream>>>(buf1);
    pack_kernel<<<NPIX, VDIM, 0, stream>>>(buf1, bf_window, Br, Bi);
    calib_pack_kernel<<<(A_PAD * VDIM + 255) / 256, 256, 0, stream>>>(calib_real, calib_imag, Ar, Ai, mAi);
    bf_gemm_kernel<<<dim3(NPIX / 16 / 8, 47, 1), 256, 0, stream>>>(Ar, Ai, mAi, Br, Bi, out);
}
